// SelfAttention_42709154792019
// MI455X (gfx1250) — matrix operand order repaired and checked by reading, not compiled, not run on hardware
//
#include <hip/hip_runtime.h>

// ---------------------------------------------------------------------------
// Problem constants (from reference)
// ---------------------------------------------------------------------------
constexpr int B    = 2;
constexpr int S    = 2048;
constexpr int DIM  = 4096;
constexpr int NH   = 32;
constexpr int NKV  = 8;
constexpr int HD   = 128;
constexpr int NREP = NH / NKV;          // 4
constexpr int TOK  = B * S;             // 4096 tokens
constexpr float SM_SCALE = 0.08838834764831845f; // 1/sqrt(128)

// Use gfx1250 async global->LDS DMA (ASYNCcnt) for tile staging.
#define ASYNC_LDS 1

// ---------------------------------------------------------------------------
// Vector types for WMMA (wave32, CDNA5)
// ---------------------------------------------------------------------------
typedef __attribute__((ext_vector_type(16))) __bf16 v16bf;
typedef __attribute__((ext_vector_type(8)))  __bf16 v8bf;
typedef __attribute__((ext_vector_type(8)))  float  v8f;

static __device__ inline v8bf ld8(const __bf16* p) { return *(const v8bf*)p; }

static __device__ inline v16bf cat16(v8bf lo, v8bf hi) {
    return __builtin_shufflevector(lo, hi, 0,1,2,3,4,5,6,7,8,9,10,11,12,13,14,15);
}

static __device__ inline v8f wmma_bf16(v16bf a, v16bf b, v8f c) {
    // D = A(16x32) * B(32x16) + C, f32 accumulate
    return __builtin_amdgcn_wmma_f32_16x16x32_bf16(false, a, false, b, (short)0, c, false, false);
}

// 16-byte async copy global -> LDS (per-lane), tracked by ASYNCcnt.
static __device__ inline void async_cp16(const __bf16* src, __bf16* dst_lds) {
#if ASYNC_LDS
    const unsigned lds_off = (unsigned)(uintptr_t)dst_lds;   // generic low 32 = LDS offset
    asm volatile("global_load_async_to_lds_b128 %0, %1, off"
                 :: "v"(lds_off), "v"(src) : "memory");
#else
    *(v8bf*)dst_lds = *(const v8bf*)src;
#endif
}

static __device__ inline void async_wait_all() {
#if ASYNC_LDS
    asm volatile("s_wait_asynccnt 0x0" ::: "memory");
#endif
}

// ---------------------------------------------------------------------------
// fp32 -> bf16 elementwise convert (grid-stride)
// ---------------------------------------------------------------------------
__global__ void convert_f32_bf16(const float* __restrict__ in, __bf16* __restrict__ out, int n) {
    for (int i = blockIdx.x * blockDim.x + threadIdx.x; i < n; i += gridDim.x * blockDim.x)
        out[i] = (__bf16)in[i];
}

// ---------------------------------------------------------------------------
// Transpose + convert weight: in fp32 [K][N] (row-major) -> out bf16 [N][K]
// ---------------------------------------------------------------------------
__global__ __launch_bounds__(256)
void transpose_convert(const float* __restrict__ in, __bf16* __restrict__ out, int K, int N) {
    __shared__ float tile[32][33];
    const int tx = threadIdx.x, ty = threadIdx.y;      // block (32,8)
    const int nb = blockIdx.x * 32, kb = blockIdx.y * 32;
    #pragma unroll
    for (int j = 0; j < 4; ++j)
        tile[ty + j * 8][tx] = in[(size_t)(kb + ty + j * 8) * N + nb + tx];
    __syncthreads();
    #pragma unroll
    for (int j = 0; j < 4; ++j)
        out[(size_t)(nb + ty + j * 8) * K + kb + tx] = (__bf16)tile[tx][ty + j * 8];
}

// ---------------------------------------------------------------------------
// Transpose V (bf16): v[b][s][kvh][d] -> vt[b][kvh][d][s]
// ---------------------------------------------------------------------------
__global__ __launch_bounds__(256)
void transpose_v(const __bf16* __restrict__ v, __bf16* __restrict__ vt) {
    __shared__ __bf16 tile[32][34];
    const int tx = threadIdx.x, ty = threadIdx.y;      // block (32,8)
    const int db = blockIdx.x * 32;
    const int sb = blockIdx.y * 32;
    const int bh = blockIdx.z;
    const int b  = bh / NKV, h = bh % NKV;
    #pragma unroll
    for (int j = 0; j < 4; ++j)
        tile[ty + j * 8][tx] = v[(((size_t)b * S + sb + ty + j * 8) * NKV + h) * HD + db + tx];
    __syncthreads();
    #pragma unroll
    for (int j = 0; j < 4; ++j)
        vt[(((size_t)b * NKV + h) * HD + db + ty + j * 8) * S + sb + tx] = tile[tx][ty + j * 8];
}

// ---------------------------------------------------------------------------
// RoPE (interleaved pairs), in-place on bf16 buffer [B][S][nh][HD]
// ---------------------------------------------------------------------------
__global__ void rope_kernel(__bf16* __restrict__ buf, const float* __restrict__ cosb,
                            const float* __restrict__ sinb, int nheads) {
    const int total = B * S * nheads * (HD / 2);
    for (int t = blockIdx.x * blockDim.x + threadIdx.x; t < total; t += gridDim.x * blockDim.x) {
        const int i = t & 63;
        const int h = (t >> 6) % nheads;
        const int s = (t / (64 * nheads)) % S;
        const int b = t / (64 * nheads * S);
        const float c  = cosb[s * 64 + i];
        const float sn = sinb[s * 64 + i];
        const size_t base = (((size_t)b * S + s) * nheads + h) * HD + 2 * i;
        const float re = (float)buf[base], im = (float)buf[base + 1];
        buf[base]     = (__bf16)(re * c - im * sn);
        buf[base + 1] = (__bf16)(re * sn + im * c);
    }
}

// ---------------------------------------------------------------------------
// bf16 GEMM: C[M][N] = A[M][K] * Bt[N][K]^T.
// 128x128 block tile, 4 waves each computing 64x64 (4x4 WMMA tiles).
// A/B tiles double-buffered in LDS via global_load_async_to_lds_b128
// (ASYNCcnt) overlapping WMMA of the current tile.
// ---------------------------------------------------------------------------
template <bool F32OUT>
__global__ __launch_bounds__(128)
void gemm_bf16_tn(const __bf16* __restrict__ A, const __bf16* __restrict__ Bt,
                  void* __restrict__ Cout, int M, int N, int K) {
    __shared__ __align__(16) __bf16 Ash[2][128][32];
    __shared__ __align__(16) __bf16 Bsh[2][128][32];

    const int tid  = threadIdx.x;
    const int lane = tid & 31;
    const int wid  = tid >> 5;
    const int lm   = lane & 15;
    const int hf   = lane >> 4;
    const int rowB = blockIdx.y * 128;
    const int colB = blockIdx.x * 128;
    const int wrow = (wid >> 1) * 64;
    const int wcol = (wid & 1) * 64;

    auto stage = [&](int buf, int k0) {
        #pragma unroll
        for (int i = 0; i < 4; ++i) {
            const int c  = tid * 4 + i;        // 0..511 chunks of 8 bf16
            const int r  = c >> 2;             // row 0..127
            const int ko = (c & 3) * 8;        // k offset 0/8/16/24
            async_cp16(A  + (size_t)(rowB + r) * K + k0 + ko, &Ash[buf][r][ko]);
            async_cp16(Bt + (size_t)(colB + r) * K + k0 + ko, &Bsh[buf][r][ko]);
        }
    };

    v8f acc[4][4];
    #pragma unroll
    for (int i = 0; i < 4; ++i)
        #pragma unroll
        for (int j = 0; j < 4; ++j)
            #pragma unroll
            for (int r = 0; r < 8; ++r) acc[i][j][r] = 0.0f;

    stage(0, 0);

    const int NT = K / 32;
    for (int t = 0; t < NT; ++t) {
        const int buf = t & 1;
        async_wait_all();                 // this wave's copies for tile t done
        __syncthreads();                  // all waves' copies visible; prior reads done
        if (t + 1 < NT) stage(buf ^ 1, (t + 1) * 32);

        v16bf af[4], bfr[4];
        #pragma unroll
        for (int i = 0; i < 4; ++i) {
            const __bf16* p = &Ash[buf][wrow + i * 16 + lm][0];
            af[i] = cat16(ld8(p + hf * 8), ld8(p + 16 + hf * 8));
        }
        #pragma unroll
        for (int j = 0; j < 4; ++j) {
            const __bf16* p = &Bsh[buf][wcol + j * 16 + lm][0];
            bfr[j] = cat16(ld8(p + hf * 8), ld8(p + 16 + hf * 8));
        }
        #pragma unroll
        for (int i = 0; i < 4; ++i)
            #pragma unroll
            for (int j = 0; j < 4; ++j)
                acc[i][j] = wmma_bf16(af[i], bfr[j], acc[i][j]);
        __syncthreads();                  // reads done before buf is re-staged
    }

    #pragma unroll
    for (int i = 0; i < 4; ++i)
        #pragma unroll
        for (int j = 0; j < 4; ++j)
            #pragma unroll
            for (int r = 0; r < 8; ++r) {
                const int m = rowB + wrow + i * 16 + hf * 8 + r;
                const int n = colB + wcol + j * 16 + lm;
                if (F32OUT) ((float*)Cout)[(size_t)m * N + n] = acc[i][j][r];
                else        ((__bf16*)Cout)[(size_t)m * N + n] = (__bf16)acc[i][j][r];
            }
}

// ---------------------------------------------------------------------------
// Flash attention (no mask). 4 waves/block share one (b,h); each wave owns 16
// queries. K/V tiles double-buffered in LDS via async DMA, shared by all
// waves. All 8 K-fragments (resp. V-fragments) are preloaded into registers
// so the ds_loads batch into one clause and WMMAs issue back-to-back.
// Row-sums come from an extra WMMA against an all-ones B fragment
// (L = P x 1), eliminating 32 ds_bpermute ops per tile; only the row-max
// butterfly uses shfl_xor.
// ---------------------------------------------------------------------------
__global__ __launch_bounds__(128)
void flash_attn(const __bf16* __restrict__ q, const __bf16* __restrict__ k,
                const __bf16* __restrict__ vt, __bf16* __restrict__ out) {
    __shared__ __align__(16) __bf16 Ksh[2][32][128];   // [key][d]
    __shared__ __align__(16) __bf16 Vsh[2][128][32];   // [d][key]
    __shared__ __align__(16) __bf16 Pl[4][16][32];     // per-wave P staging

    const int tid  = threadIdx.x;
    const int lane = tid & 31;
    const int wid  = tid >> 5;
    const int lm   = lane & 15;
    const int hf   = lane >> 4;

    const int bh  = blockIdx.y;
    const int b   = bh / NH;
    const int h   = bh % NH;
    const int kvh = h / NREP;
    const int q0  = blockIdx.x * 64 + wid * 16;

    const __bf16* kbase  = k  + ((size_t)b * S * NKV + kvh) * HD;     // +key*NKV*HD
    const __bf16* vtbase = vt + (((size_t)b * NKV + kvh) * HD) * S;   // +d*S + key

    auto stage = [&](int buf, int kt) {
        #pragma unroll
        for (int i = 0; i < 4; ++i) {          // K tile: 512 chunks
            const int c   = tid * 4 + i;
            const int key = c >> 4;            // 0..31
            const int d   = (c & 15) * 8;      // 0..120
            async_cp16(kbase + (size_t)(kt + key) * (NKV * HD) + d, &Ksh[buf][key][d]);
        }
        #pragma unroll
        for (int i = 0; i < 4; ++i) {          // V tile: 512 chunks
            const int c   = tid * 4 + i;
            const int d   = c >> 2;            // 0..127
            const int key = (c & 3) * 8;       // 0..24
            async_cp16(vtbase + (size_t)d * S + kt + key, &Vsh[buf][d][key]);
        }
    };

    // Q fragments: rows q0+lm, full 128-d as 4 k-steps of 32 (held in regs)
    const __bf16* qbase = q + (((size_t)b * S + q0 + lm) * NH + h) * HD;
    v16bf qf[4];
    #pragma unroll
    for (int kk = 0; kk < 4; ++kk) {
        const __bf16* p = qbase + kk * 32 + hf * 8;
        qf[kk] = cat16(ld8(p), ld8(p + 16));
    }

    // all-ones B fragment for row-sum WMMA (any layout: all elements equal)
    v16bf onesf;
    #pragma unroll
    for (int e = 0; e < 16; ++e) onesf[e] = (__bf16)1.0f;

    float mrow[8], lrow[8];
    v8f   oacc[8];
    #pragma unroll
    for (int r = 0; r < 8; ++r) { mrow[r] = -3.0e38f; lrow[r] = 0.0f; }
    #pragma unroll
    for (int j = 0; j < 8; ++j)
        #pragma unroll
        for (int r = 0; r < 8; ++r) oacc[j][r] = 0.0f;

    stage(0, 0);

    for (int t = 0; t < S / 32; ++t) {
        const int buf = t & 1;
        const int kt  = t * 32;
        async_wait_all();
        __syncthreads();
        if (kt + 32 < S) stage(buf ^ 1, kt + 32);

        // ---- preload all K fragments, then batch the 8 score WMMAs ----
        v16bf kf[8];
        #pragma unroll
        for (int kk = 0; kk < 4; ++kk) {
            const __bf16* p0 = &Ksh[buf][lm][kk * 32 + hf * 8];
            const __bf16* p1 = &Ksh[buf][16 + lm][kk * 32 + hf * 8];
            kf[kk * 2]     = cat16(ld8(p0), ld8(p0 + 16));
            kf[kk * 2 + 1] = cat16(ld8(p1), ld8(p1 + 16));
        }
        v8f s0, s1;
        #pragma unroll
        for (int r = 0; r < 8; ++r) { s0[r] = 0.0f; s1[r] = 0.0f; }
        #pragma unroll
        for (int kk = 0; kk < 4; ++kk) {
            s0 = wmma_bf16(qf[kk], kf[kk * 2],     s0);
            s1 = wmma_bf16(qf[kk], kf[kk * 2 + 1], s1);
        }

        // ---- online softmax: row-max butterfly across 16-lane groups ----
        float pm[8];
        #pragma unroll
        for (int r = 0; r < 8; ++r) pm[r] = fmaxf(s0[r], s1[r]);
        #pragma unroll
        for (int msk = 1; msk <= 8; msk <<= 1)
            #pragma unroll
            for (int r = 0; r < 8; ++r) pm[r] = fmaxf(pm[r], __shfl_xor(pm[r], msk, 32));

        float alpha[8], p0v[8], p1v[8];
        #pragma unroll
        for (int r = 0; r < 8; ++r) {
            const float mn = fmaxf(mrow[r], pm[r]);
            alpha[r] = __expf(SM_SCALE * (mrow[r] - mn));
            mrow[r]  = mn;
            p0v[r] = __expf(SM_SCALE * (s0[r] - mn));
            p1v[r] = __expf(SM_SCALE * (s1[r] - mn));
        }

        // ---- stage P via wave-private LDS into A-fragment layout ----
        #pragma unroll
        for (int r = 0; r < 8; ++r) {
            Pl[wid][hf * 8 + r][lm]      = (__bf16)p0v[r];
            Pl[wid][hf * 8 + r][lm + 16] = (__bf16)p1v[r];
        }
        v16bf pa = cat16(ld8(&Pl[wid][lm][hf * 8]), ld8(&Pl[wid][lm][16 + hf * 8]));

        // ---- row sums via WMMA: L = P x ones (every C column = rowsum) ----
        v8f zf;
        #pragma unroll
        for (int r = 0; r < 8; ++r) zf[r] = 0.0f;
        v8f lsum = wmma_bf16(pa, onesf, zf);
        #pragma unroll
        for (int r = 0; r < 8; ++r) lrow[r] = lrow[r] * alpha[r] + lsum[r];

        // ---- preload all V fragments, rescale accumulators, batch PV ----
        v16bf vf[8];
        #pragma unroll
        for (int j = 0; j < 8; ++j) {
            const __bf16* vp = &Vsh[buf][j * 16 + lm][hf * 8];
            vf[j] = cat16(ld8(vp), ld8(vp + 16));
        }
        #pragma unroll
        for (int j = 0; j < 8; ++j)
            #pragma unroll
            for (int r = 0; r < 8; ++r) oacc[j][r] *= alpha[r];
        #pragma unroll
        for (int j = 0; j < 8; ++j)
            oacc[j] = wmma_bf16(pa, vf[j], oacc[j]);
        __syncthreads();      // tile reads done before re-staging this buffer
    }

    // ---- normalize + write [b][s][h][d] bf16 ----
    __bf16* ob = out + (((size_t)b * S + q0) * NH + h) * HD;
    #pragma unroll
    for (int r = 0; r < 8; ++r) {
        const float inv = 1.0f / lrow[r];
        const int   m   = hf * 8 + r;
        #pragma unroll
        for (int j = 0; j < 8; ++j)
            ob[(size_t)m * (NH * HD) + j * 16 + lm] = (__bf16)(oacc[j][r] * inv);
    }
}

// ---------------------------------------------------------------------------
// Host-side orchestration
// ---------------------------------------------------------------------------
extern "C" void kernel_launch(void* const* d_in, const int* in_sizes, int n_in,
                              void* d_out, int out_size, void* d_ws, size_t ws_size,
                              hipStream_t stream) {
    (void)in_sizes; (void)n_in; (void)out_size; (void)ws_size;
    const float* x    = (const float*)d_in[0];
    const float* wq   = (const float*)d_in[1];
    const float* wk   = (const float*)d_in[2];
    const float* wv   = (const float*)d_in[3];
    const float* wo   = (const float*)d_in[4];
    const float* cosb = (const float*)d_in[7];
    const float* sinb = (const float*)d_in[8];
    // d_in[5]/[6] (caches) and d_in[9] (start_pos==0) are not needed.

    char* ws = (char*)d_ws;
    __bf16* xb  = (__bf16*)ws;                 ws += (size_t)TOK * DIM * 2;
    __bf16* wqt = (__bf16*)ws;                 ws += (size_t)DIM * (NH * HD) * 2;
    __bf16* wkt = (__bf16*)ws;                 ws += (size_t)DIM * (NKV * HD) * 2;
    __bf16* wvt = (__bf16*)ws;                 ws += (size_t)DIM * (NKV * HD) * 2;
    __bf16* wot = (__bf16*)ws;                 ws += (size_t)(NH * HD) * DIM * 2;
    __bf16* qb  = (__bf16*)ws;                 ws += (size_t)TOK * NH * HD * 2;
    __bf16* kb  = (__bf16*)ws;                 ws += (size_t)TOK * NKV * HD * 2;
    __bf16* vb  = (__bf16*)ws;                 ws += (size_t)TOK * NKV * HD * 2;
    __bf16* vtb = (__bf16*)ws;                 ws += (size_t)TOK * NKV * HD * 2;
    __bf16* ab  = (__bf16*)ws;

    convert_f32_bf16<<<2048, 256, 0, stream>>>(x, xb, TOK * DIM);
    transpose_convert<<<dim3((NH * HD) / 32, DIM / 32), dim3(32, 8), 0, stream>>>(wq, wqt, DIM, NH * HD);
    transpose_convert<<<dim3((NKV * HD) / 32, DIM / 32), dim3(32, 8), 0, stream>>>(wk, wkt, DIM, NKV * HD);
    transpose_convert<<<dim3((NKV * HD) / 32, DIM / 32), dim3(32, 8), 0, stream>>>(wv, wvt, DIM, NKV * HD);
    transpose_convert<<<dim3(DIM / 32, (NH * HD) / 32), dim3(32, 8), 0, stream>>>(wo, wot, NH * HD, DIM);

    gemm_bf16_tn<false><<<dim3((NH * HD) / 128, TOK / 128), 128, 0, stream>>>(xb, wqt, qb, TOK, NH * HD, DIM);
    gemm_bf16_tn<false><<<dim3((NKV * HD) / 128, TOK / 128), 128, 0, stream>>>(xb, wkt, kb, TOK, NKV * HD, DIM);
    gemm_bf16_tn<false><<<dim3((NKV * HD) / 128, TOK / 128), 128, 0, stream>>>(xb, wvt, vb, TOK, NKV * HD, DIM);

    rope_kernel<<<8192, 256, 0, stream>>>(qb, cosb, sinb, NH);
    rope_kernel<<<2048, 256, 0, stream>>>(kb, cosb, sinb, NKV);

    transpose_v<<<dim3(HD / 32, S / 32, B * NKV), dim3(32, 8), 0, stream>>>(vb, vtb);

    flash_attn<<<dim3(S / 64, B * NH), 128, 0, stream>>>(qb, kb, vtb, ab);

    gemm_bf16_tn<true><<<dim3(DIM / 128, TOK / 128), 128, 0, stream>>>(ab, wot, d_out, TOK, DIM, NH * HD);
}
